// MultiHeadAttention_52304111731071
// MI455X (gfx1250) — matrix operand order repaired and checked by reading, not compiled, not run on hardware
//
#include <hip/hip_runtime.h>

typedef __bf16 bf16;
typedef bf16  v16bf __attribute__((ext_vector_type(16)));
typedef bf16  v8bf  __attribute__((ext_vector_type(8)));
typedef float v8f   __attribute__((ext_vector_type(8)));

#define B_  4
#define C_  1024
#define T_  1024
#define H_  16
#define DH_ 64

static __device__ __forceinline__ v16bf cat8(v8bf lo, v8bf hi) {
    return __builtin_shufflevector(lo, hi, 0,1,2,3,4,5,6,7,8,9,10,11,12,13,14,15);
}

// ---------------------------------------------------------------------------
// f32 (b, R, Ccols) -> bf16 (b, Ccols, R)   (tiled transpose + convert)
// ---------------------------------------------------------------------------
__global__ void transpose_cvt_bf16(const float* __restrict__ X, bf16* __restrict__ XT,
                                   int R, int Ccols)
{
    __shared__ bf16 tile[32][33];
    const int b  = blockIdx.z;
    const int r0 = blockIdx.y * 32;
    const int c0 = blockIdx.x * 32;
    const float* x = X + (size_t)b * R * Ccols;
    bf16* xt = XT + (size_t)b * R * Ccols;
    const int tx = threadIdx.x, ty = threadIdx.y;   // 32 x 8
#pragma unroll
    for (int i = 0; i < 32; i += 8)
        tile[ty + i][tx] = (bf16)x[(size_t)(r0 + ty + i) * Ccols + c0 + tx];
    __syncthreads();
#pragma unroll
    for (int i = 0; i < 32; i += 8)
        xt[(size_t)(c0 + ty + i) * R + r0 + tx] = tile[tx][ty + i];
}

// ---------------------------------------------------------------------------
// flat f32 -> bf16 convert (weights)
// ---------------------------------------------------------------------------
__global__ void cvt_bf16(const float* __restrict__ src, bf16* __restrict__ dst, int n)
{
    int i = blockIdx.x * blockDim.x + threadIdx.x;
    if (i < n) dst[i] = (bf16)src[i];
}

// ---------------------------------------------------------------------------
// GEMM: Y[b] (MxN, f32, row-major) = W (MxK bf16 row-major) * X[b] + bias
// X given transposed per batch: XT[b] is N x K row-major (K contiguous).
// block = 256 threads (8 waves, 4x2), block tile 128(M) x 64(N),
// wave tile 32x32 = 2x2 WMMA fragments, K step 32.
// ---------------------------------------------------------------------------
__global__ void gemm_wmma_bf16(const bf16* __restrict__ W, const bf16* __restrict__ XT,
                               const float* __restrict__ bias, float* __restrict__ Y,
                               int M, int N, int K)
{
    const int b = blockIdx.z;
    const bf16* xt = XT + (size_t)b * N * K;
    float* y = Y + (size_t)b * M * N;

    const int lane = threadIdx.x & 31;
    const int wave = threadIdx.x >> 5;      // 0..7
    const int wm   = wave >> 1;             // 0..3
    const int wn   = wave & 1;              // 0..1
    const int m0   = blockIdx.y * 128 + wm * 32;
    const int n0   = blockIdx.x * 64  + wn * 32;
    const int l16  = lane & 15;
    const int hh   = lane >> 4;

    v8f acc[2][2];
#pragma unroll
    for (int i = 0; i < 2; i++)
#pragma unroll
        for (int j = 0; j < 2; j++)
#pragma unroll
            for (int r = 0; r < 8; r++) acc[i][j][r] = 0.f;

    for (int k0 = 0; k0 < K; k0 += 32) {
        v16bf afrag[2], bfrag[2];
#pragma unroll
        for (int i = 0; i < 2; i++) {
            const bf16* ap = W + (size_t)(m0 + i * 16 + l16) * K + k0;
            afrag[i] = cat8(*(const v8bf*)(ap + 16 * hh),
                            *(const v8bf*)(ap + 16 * hh + 8));
        }
#pragma unroll
        for (int j = 0; j < 2; j++) {
            const bf16* bp = xt + (size_t)(n0 + j * 16 + l16) * K + k0 + 16 * hh;
            bfrag[j] = cat8(*(const v8bf*)(bp), *(const v8bf*)(bp + 8));
        }
#pragma unroll
        for (int i = 0; i < 2; i++)
#pragma unroll
            for (int j = 0; j < 2; j++)
                acc[i][j] = __builtin_amdgcn_wmma_f32_16x16x32_bf16(
                    false, afrag[i], false, bfrag[j], (short)0, acc[i][j], false, false);
    }

#pragma unroll
    for (int i = 0; i < 2; i++) {
#pragma unroll
        for (int r = 0; r < 8; r++) {
            const int m = m0 + i * 16 + 8 * hh + r;
            const float bv = bias[m];
#pragma unroll
            for (int j = 0; j < 2; j++)
                y[(size_t)m * N + n0 + j * 16 + l16] = acc[i][j][r] + bv;
        }
    }
}

// ---------------------------------------------------------------------------
// RoPE (first 32 dims) + scale + convert: f32 (b, ch=h*64+d, t) -> bf16 (b,h,t,d)
// ---------------------------------------------------------------------------
__global__ void rope_cvt(const float* __restrict__ Z, bf16* __restrict__ Out,
                         float scale, int srcBatchStride)
{
    const int idx = blockIdx.x * blockDim.x + threadIdx.x;   // B*H*T*DH exact
    const int d = idx & 63;
    const int t = (idx >> 6) & 1023;
    const int h = (idx >> 16) & 15;
    const int b = idx >> 20;
    const float* z = Z + (size_t)b * srcBatchStride;
    float v = z[(size_t)(h * DH_ + d) * T_ + t];
    if (d < 32) {
        const int i = d & 15;
        const float ang = (float)t * __powf(10000.f, -(float)i * (1.f / 16.f));
        float cs, sn;
        __sincosf(ang, &sn, &cs);
        const int dp = (d < 16) ? d + 16 : d - 16;
        const float v2 = z[(size_t)(h * DH_ + dp) * T_ + t];
        v = (d < 16) ? (v * cs - v2 * sn) : (v * cs + v2 * sn);
    }
    Out[idx] = (bf16)(v * scale);
}

// V: f32 kv (b, 2C, t) channels [C,2C) -> bf16 (B,H,DH,T) == straight convert,
// fully coalesced on both sides (channel ordering (h,d) == kv channel order).
__global__ void v_cvt(const float* __restrict__ KV, bf16* __restrict__ Out)
{
    const int idx = blockIdx.x * blockDim.x + threadIdx.x;   // B*C*T exact
    const int b   = idx >> 20;                               // C_*T_ == 1<<20
    const int rem = idx & ((1 << 20) - 1);
    Out[idx] = (bf16)KV[(size_t)b * 2 * C_ * T_ + (size_t)C_ * T_ + rem];
}

// ---------------------------------------------------------------------------
// Flash attention: Q,K bf16 (B,H,T,DH); V bf16 (B,H,DH,T);
// output transposed bf16 (B,T,C).
// block = 128 threads (4 independent waves; no cross-wave LDS sharing).
// Wave owns 16 query rows; kv chunks of 32.
// Q pre-scaled by 1/sqrt(DH) (folded into rope_cvt).
// ---------------------------------------------------------------------------
__global__ void attn_wmma(const bf16* __restrict__ Q, const bf16* __restrict__ Km,
                          const bf16* __restrict__ Vm, bf16* __restrict__ OutT)
{
    __shared__ alignas(16) bf16 Plds[4][16][40];     // per-wave P relayout scratch

    const int b = blockIdx.z, h = blockIdx.y;
    const int wave = threadIdx.x >> 5;
    const int lane = threadIdx.x & 31;
    const int l16 = lane & 15, hh = lane >> 4;
    const int q0 = blockIdx.x * 64 + wave * 16;

    const size_t headOff = ((size_t)b * H_ + h) * T_ * DH_;
    const bf16* qb = Q + headOff;
    const bf16* kb = Km + headOff;
    const bf16* vb = Vm + headOff;                   // (DH, T) layout per head

    v16bf qA[2];
#pragma unroll
    for (int dc = 0; dc < 2; dc++) {
        const bf16* qp = qb + (size_t)(q0 + l16) * DH_ + dc * 32;
        qA[dc] = cat8(*(const v8bf*)(qp + 16 * hh), *(const v8bf*)(qp + 16 * hh + 8));
    }

    float mrow[8], lrow[8];
#pragma unroll
    for (int r = 0; r < 8; r++) { mrow[r] = -1e30f; lrow[r] = 0.f; }
    v8f acc[4];
#pragma unroll
    for (int j = 0; j < 4; j++)
#pragma unroll
        for (int r = 0; r < 8; r++) acc[j][r] = 0.f;

    for (int kj = 0; kj < T_; kj += 32) {
        // scores S = Q * K^T  (16 x 32, two 16-col fragments, K-dim = 64)
        v8f s[2];
#pragma unroll
        for (int ns = 0; ns < 2; ns++) {
            v8f cf;
#pragma unroll
            for (int r = 0; r < 8; r++) cf[r] = 0.f;
#pragma unroll
            for (int dc = 0; dc < 2; dc++) {
                const bf16* kp = kb + (size_t)(kj + ns * 16 + l16) * DH_ + dc * 32 + 16 * hh;
                v16bf kB = cat8(*(const v8bf*)kp, *(const v8bf*)(kp + 8));
                cf = __builtin_amdgcn_wmma_f32_16x16x32_bf16(
                        false, qA[dc], false, kB, (short)0, cf, false, false);
            }
            s[ns] = cf;
        }

        // online softmax (row m = 8*hh + r lives on the 16 lanes of half hh)
        float alpha[8];
#pragma unroll
        for (int r = 0; r < 8; r++) {
            float v = fmaxf(s[0][r], s[1][r]);
            v = fmaxf(v, __shfl_xor(v, 1));
            v = fmaxf(v, __shfl_xor(v, 2));
            v = fmaxf(v, __shfl_xor(v, 4));
            v = fmaxf(v, __shfl_xor(v, 8));
            const float mnew = fmaxf(mrow[r], v);
            alpha[r] = __expf(mrow[r] - mnew);
            const float p0 = __expf(s[0][r] - mnew);
            const float p1 = __expf(s[1][r] - mnew);
            s[0][r] = p0; s[1][r] = p1;
            float rs = p0 + p1;
            rs += __shfl_xor(rs, 1);
            rs += __shfl_xor(rs, 2);
            rs += __shfl_xor(rs, 4);
            rs += __shfl_xor(rs, 8);
            lrow[r] = lrow[r] * alpha[r] + rs;
            mrow[r] = mnew;
        }
#pragma unroll
        for (int j = 0; j < 4; j++)
#pragma unroll
            for (int r = 0; r < 8; r++) acc[j][r] *= alpha[r];

        // C-fragment -> A-fragment relayout of P through per-wave LDS
        // (same-wave DS ops are in-order; no barrier needed)
#pragma unroll
        for (int ns = 0; ns < 2; ns++)
#pragma unroll
            for (int r = 0; r < 8; r++)
                Plds[wave][8 * hh + r][ns * 16 + l16] = (bf16)s[ns][r];

        const bf16* pp = &Plds[wave][l16][0];
        v16bf pA = cat8(*(const v8bf*)(pp + 16 * hh), *(const v8bf*)(pp + 16 * hh + 8));

        // P @ V : V fragments straight from global ((DH,T): kv contiguous)
#pragma unroll
        for (int j = 0; j < 4; j++) {
            const bf16* vp = vb + (size_t)(j * 16 + l16) * T_ + kj + 16 * hh;
            v16bf vB = cat8(*(const v8bf*)vp, *(const v8bf*)(vp + 8));
            acc[j] = __builtin_amdgcn_wmma_f32_16x16x32_bf16(
                        false, pA, false, vB, (short)0, acc[j], false, false);
        }
    }

    // epilogue: normalize, store transposed: OutT[b][t][h*64 + d]
#pragma unroll
    for (int r = 0; r < 8; r++) {
        const float inv = 1.f / lrow[r];
        const int t = q0 + 8 * hh + r;
#pragma unroll
        for (int j = 0; j < 4; j++)
            OutT[((size_t)b * T_ + t) * C_ + h * DH_ + j * 16 + l16] =
                (bf16)(acc[j][r] * inv);
    }
}

// ---------------------------------------------------------------------------
extern "C" void kernel_launch(void* const* d_in, const int* in_sizes, int n_in,
                              void* d_out, int out_size, void* d_ws, size_t ws_size,
                              hipStream_t stream)
{
    const float* x    = (const float*)d_in[0];
    const float* c    = (const float*)d_in[1];
    const float* q_w  = (const float*)d_in[2];
    const float* q_b  = (const float*)d_in[3];
    const float* kv_w = (const float*)d_in[4];
    const float* kv_b = (const float*)d_in[5];
    const float* o_w  = (const float*)d_in[6];
    const float* o_b  = (const float*)d_in[7];
    float* out = (float*)d_out;

    const size_t NBCT = (size_t)B_ * C_ * T_;          // 4M elements
    size_t off = 0;
    char* ws = (char*)d_ws;
    auto take = [&](size_t bytes) -> char* {
        char* p = ws + off;
        off += (bytes + 255) & ~(size_t)255;
        return p;
    };
    bf16*  xT   = (bf16*)take(NBCT * 2);               // (B,T,C)
    bf16*  cT   = (bf16*)take(NBCT * 2);               // (B,T,C)
    bf16*  qwb  = (bf16*)take((size_t)C_ * C_ * 2);
    bf16*  kvwb = (bf16*)take((size_t)2 * C_ * C_ * 2);
    bf16*  owb  = (bf16*)take((size_t)C_ * C_ * 2);
    float* qf   = (float*)take(NBCT * 4);              // (B,C,T)
    float* kvf  = (float*)take(2 * NBCT * 4);          // (B,2C,T)
    bf16*  qbf  = (bf16*)take(NBCT * 2);               // (B,H,T,DH)
    bf16*  kbf  = (bf16*)take(NBCT * 2);               // (B,H,T,DH)
    bf16*  vbf  = (bf16*)take(NBCT * 2);               // (B,H,DH,T)
    bf16*  aT   = (bf16*)take(NBCT * 2);               // attn out (B,T,C)

    // 1) activations -> bf16, transposed (B,T,C)
    {
        dim3 g(T_ / 32, C_ / 32, B_), blk(32, 8);
        transpose_cvt_bf16<<<g, blk, 0, stream>>>(x, xT, C_, T_);
        transpose_cvt_bf16<<<g, blk, 0, stream>>>(c, cT, C_, T_);
    }
    // 2) weights -> bf16
    cvt_bf16<<<(C_ * C_) / 256, 256, 0, stream>>>(q_w, qwb, C_ * C_);
    cvt_bf16<<<(2 * C_ * C_) / 256, 256, 0, stream>>>(kv_w, kvwb, 2 * C_ * C_);
    cvt_bf16<<<(C_ * C_) / 256, 256, 0, stream>>>(o_w, owb, C_ * C_);

    // 3) projections (WMMA bf16, f32 accum)
    gemm_wmma_bf16<<<dim3(T_ / 64, C_ / 128, B_), 256, 0, stream>>>(
        qwb, xT, q_b, qf, C_, T_, C_);
    gemm_wmma_bf16<<<dim3(T_ / 64, (2 * C_) / 128, B_), 256, 0, stream>>>(
        kvwb, cT, kv_b, kvf, 2 * C_, T_, C_);

    // 4) RoPE + layout; fold 1/sqrt(DH) into Q
    {
        const int total = B_ * H_ * T_ * DH_;          // == 4M
        rope_cvt<<<total / 256, 256, 0, stream>>>(qf, qbf, 0.125f, C_ * T_);
        rope_cvt<<<total / 256, 256, 0, stream>>>(kvf, kbf, 1.0f, 2 * C_ * T_);
        v_cvt<<<total / 256, 256, 0, stream>>>(kvf, vbf);
    }

    // 5) attention -> aT (B,T,C)
    attn_wmma<<<dim3(T_ / 64, H_, B_), 128, 0, stream>>>(qbf, kbf, vbf, aT);

    // 6) output projection -> d_out (B,C,T) f32
    gemm_wmma_bf16<<<dim3(T_ / 64, C_ / 128, B_), 256, 0, stream>>>(
        owb, aT, o_b, out, C_, T_, C_);
}
